// MultiHeadAttention_24550033064050
// MI455X (gfx1250) — matrix operand order repaired and checked by reading, not compiled, not run on hardware
//
#include <hip/hip_runtime.h>

// Problem constants (reference: B=4, S=2048, D=1024, H=16, HD=64)
#define S_LEN  2048
#define DMODEL 1024
#define NHEAD  16
#define HDIM   64
#define MROWS  8192   // B*S

typedef __attribute__((ext_vector_type(16))) __bf16 bf16x16;
typedef __attribute__((ext_vector_type(8)))  float  f32x8;

__device__ __forceinline__ f32x8 wmma_bf16(bf16x16 a, bf16x16 b, f32x8 c) {
  // D = A(16x32 bf16) x B(32x16 bf16) + C(16x16 f32)
  return __builtin_amdgcn_wmma_f32_16x16x32_bf16(false, a, false, b, (short)0, c, false, false);
}

// ---- Fragment loaders (CDNA5 wave32 WMMA layouts, CDNA5 ISA) ----
// A (16x32 bf16): lane L -> row = L&15; elements e<8 -> K = (L>>4)*8+e,
//                 e>=8 -> K = 16+(L>>4)*8+(e-8)
__device__ __forceinline__ bf16x16 frag_a_bf16(const __bf16* __restrict__ base,
                                               int ld, int lane) {
  const int g = lane >> 4;
  const __bf16* row = base + (size_t)(lane & 15) * ld;
  bf16x16 a;
#pragma unroll
  for (int e = 0; e < 8; ++e) a[e] = row[g * 8 + e];
#pragma unroll
  for (int e = 0; e < 8; ++e) a[e + 8] = row[16 + g * 8 + e];
  return a;
}

__device__ __forceinline__ bf16x16 frag_a_f32(const float* __restrict__ base,
                                              int ld, int lane) {
  const int g = lane >> 4;
  const float* row = base + (size_t)(lane & 15) * ld;
  bf16x16 a;
#pragma unroll
  for (int e = 0; e < 8; ++e) a[e] = (__bf16)row[g * 8 + e];
#pragma unroll
  for (int e = 0; e < 8; ++e) a[e + 8] = (__bf16)row[16 + g * 8 + e];
  return a;
}

// B (32x16 bf16) where B[kk][n] = src[n*ld + kk] (src rows are the N columns):
// lane L -> N = L&15; elements e<8 -> K = (L>>4)*8+e, e>=8 -> K = 16+(L>>4)*8+(e-8)
__device__ __forceinline__ bf16x16 frag_b_rows_bf16(const __bf16* __restrict__ base,
                                                    int ld, int lane) {
  const int g = lane >> 4;
  const __bf16* row = base + (size_t)(lane & 15) * ld;
  bf16x16 b;
#pragma unroll
  for (int e = 0; e < 16; ++e) b[e] = row[(e < 8) ? (g * 8 + e) : (16 + g * 8 + e - 8)];
  return b;
}

// ============================================================================
// Kernel 0: one-shot f32 -> bf16 weight conversion (weights re-read 256x by
// the GEMMs, so converting once halves the dominant inner-loop operand bytes)
// ============================================================================
__global__ __launch_bounds__(256) void cvt_w_k(const float* __restrict__ src,
                                               __bf16* __restrict__ dst, int n) {
  const int i = (blockIdx.x * 256 + threadIdx.x) * 4;
  if (i < n) {
    const float4 v = *(const float4*)(src + i);
    dst[i + 0] = (__bf16)v.x;
    dst[i + 1] = (__bf16)v.y;
    dst[i + 2] = (__bf16)v.z;
    dst[i + 3] = (__bf16)v.w;
  }
}

// ============================================================================
// Kernel 1: projection GEMM  Y(bf16)[M,1024] = X[M,1024] @ Wb^T + bias
// Block: 256 thr = 8 waves (2 M-waves x 4 N-waves). Wave tile: 16 x 64.
// Block tile: 32 x 256. Grid: (1024/256, 8192/32)
// ============================================================================
__global__ __launch_bounds__(256) void proj_gemm_k(const float* __restrict__ X,
                                                   const __bf16* __restrict__ Wb,
                                                   const float* __restrict__ bias,
                                                   __bf16* __restrict__ Y) {
  const int lane = threadIdx.x & 31;
  const int wave = threadIdx.x >> 5;
  const int m0 = blockIdx.y * 32 + (wave >> 2) * 16;
  const int n0 = blockIdx.x * 256 + (wave & 3) * 64;

  f32x8 acc[4] = {};
  for (int kk = 0; kk < DMODEL; kk += 32) {
    bf16x16 a = frag_a_f32(X + (size_t)m0 * DMODEL + kk, DMODEL, lane);
#pragma unroll
    for (int t = 0; t < 4; ++t) {
      bf16x16 b = frag_b_rows_bf16(Wb + (size_t)(n0 + t * 16) * DMODEL + kk, DMODEL, lane);
      acc[t] = wmma_bf16(a, b, acc[t]);
    }
  }
  const int g = lane >> 4, ln = lane & 15;
#pragma unroll
  for (int t = 0; t < 4; ++t) {
    const int col = n0 + t * 16 + ln;
    const float bv = bias[col];
#pragma unroll
    for (int r = 0; r < 8; ++r) {
      const int row = m0 + r + 8 * g;
      Y[(size_t)row * DMODEL + col] = (__bf16)(acc[t][r] + bv);
    }
  }
}

// ============================================================================
// Kernel 2: causal flash attention per (b,h) over contiguous [S,64] slabs.
// Block: 8 waves, each owns a 16-row q tile (block covers 128 q rows).
// Double-buffered K staging via GLOBAL_LOAD_ASYNC_TO_LDS_B128 (ASYNCcnt):
// chunk i+1's DMA is in flight while chunk i's WMMAs run.
// V staged transposed (2-byte scatter); P transposed via per-wave LDS.
// Grid: (S/128, H, B)
// ============================================================================
__global__ __launch_bounds__(256) void attn_flash_k(const __bf16* __restrict__ Qp,
                                                    const __bf16* __restrict__ Kp,
                                                    const __bf16* __restrict__ Vp,
                                                    const int* __restrict__ pad,
                                                    __bf16* __restrict__ O) {
  __shared__ __bf16 Kl[2][32][64];    // K chunk double buffer (async DMA target)
  __shared__ __bf16 Vt[2][64][34];    // V chunk transposed double buffer (+pad)
  __shared__ __bf16 Pl[8][16][32];    // per-wave probability tile

  const int lane = threadIdx.x & 31;
  const int wave = threadIdx.x >> 5;
  const int g = lane >> 4, ln = lane & 15;
  const int b = blockIdx.z, h = blockIdx.y;
  const int q0 = blockIdx.x * 128 + wave * 16;
  const int qhi = q0 + 15;
  const size_t headoff = ((size_t)b * NHEAD + h) * (size_t)S_LEN * HDIM;
  const __bf16* Qh = Qp + headoff;
  const __bf16* Kh = Kp + headoff;
  const __bf16* Vh = Vp + headoff;
  const int* padb = pad + b * S_LEN;

  // Q A-fragments resident in VGPRs for the whole k loop
  bf16x16 qa0 = frag_a_bf16(Qh + (size_t)q0 * HDIM + 0, HDIM, lane);
  bf16x16 qa1 = frag_a_bf16(Qh + (size_t)q0 * HDIM + 32, HDIM, lane);

  f32x8 o[4] = {};
  float mrow[8], lrow[8];
#pragma unroll
  for (int r = 0; r < 8; ++r) { mrow[r] = -3.0e38f; lrow[r] = 0.0f; }

  const int nchunks = (blockIdx.x * 128 + 128) / 32;  // causal bound for block
  const int srow = threadIdx.x >> 3;                  // staging row 0..31
  const int sseg = (threadIdx.x & 7) * 8;             // staging column segment
  const int vkrow = (threadIdx.x * 8) >> 6;           // V staging source row
  const int vcol0 = (threadIdx.x * 8) & 63;           // V staging source col

  // ---- prologue: stage chunk 0 into buffer 0 ----
  {
    const unsigned lds_k = (unsigned)(size_t)(void*)&Kl[0][srow][sseg];
    const unsigned long long g_k =
        (unsigned long long)(size_t)(const void*)(Kh + (size_t)srow * HDIM + sseg);
    asm volatile("global_load_async_to_lds_b128 %0, %1, off"
                 :: "v"(lds_k), "v"(g_k) : "memory");
    const __bf16* src = Vh + (size_t)vkrow * HDIM + vcol0;
#pragma unroll
    for (int i = 0; i < 8; ++i) Vt[0][vcol0 + i][vkrow] = src[i];
  }

  for (int ic = 0; ic < nchunks; ++ic) {
    const int kc = ic * 32;
    const int p = ic & 1;
    __syncthreads();   // everyone done reading buffer 1-p from iteration ic-1
    if (ic + 1 < nchunks) {
      // stage chunk ic+1 into buffer 1-p while chunk ic is consumed
      const int kn = kc + 32;
      const unsigned lds_k = (unsigned)(size_t)(void*)&Kl[1 - p][srow][sseg];
      const unsigned long long g_k =
          (unsigned long long)(size_t)(const void*)(Kh + (size_t)(kn + srow) * HDIM + sseg);
      asm volatile("global_load_async_to_lds_b128 %0, %1, off"
                   :: "v"(lds_k), "v"(g_k) : "memory");
      const __bf16* src = Vh + (size_t)(kn + vkrow) * HDIM + vcol0;
#pragma unroll
      for (int i = 0; i < 8; ++i) Vt[1 - p][vcol0 + i][vkrow] = src[i];
      if (kn + 32 < nchunks * 32) {   // prefetch chunk ic+2 (global_prefetch_b8)
        __builtin_prefetch(Kh + (size_t)(kn + 32 + srow) * HDIM + sseg, 0, 1);
        __builtin_prefetch(Vh + (size_t)(kn + 32 + srow) * HDIM + sseg, 0, 1);
      }
      // in-order async retirement: <=1 outstanding means chunk ic has landed
      asm volatile("s_wait_asynccnt 0x1" ::: "memory");
    } else {
      asm volatile("s_wait_asynccnt 0x0" ::: "memory");
    }
    __syncthreads();   // chunk ic visible block-wide; chunk ic+1 DMA in flight
    if (kc > qhi) continue;   // causal: this wave's rows never see these keys

    // scores: two 16x16 tiles, each accumulated over HD=64 (2 WMMAs)
    f32x8 s[2] = {};
#pragma unroll
    for (int t = 0; t < 2; ++t) {
      bf16x16 kb0 = frag_b_rows_bf16(&Kl[p][t * 16][0], 64, lane);
      s[t] = wmma_bf16(qa0, kb0, s[t]);
      bf16x16 kb1 = frag_b_rows_bf16(&Kl[p][t * 16][32], 64, lane);
      s[t] = wmma_bf16(qa1, kb1, s[t]);
    }

    // mask (pad + causal), scale 1/8, online softmax (width-16 shuffles)
    const int kcol0 = kc + ln, kcol1 = kc + 16 + ln;
    const bool pm0 = padb[kcol0] != 0;
    const bool pm1 = padb[kcol1] != 0;
#pragma unroll
    for (int r = 0; r < 8; ++r) {
      const int qrow = q0 + r + 8 * g;
      float v0 = (pm0 && kcol0 <= qrow) ? s[0][r] * 0.125f : -3.0e38f;
      float v1 = (pm1 && kcol1 <= qrow) ? s[1][r] * 0.125f : -3.0e38f;
      float mx = fmaxf(v0, v1);
#pragma unroll
      for (int off = 1; off < 16; off <<= 1) mx = fmaxf(mx, __shfl_xor(mx, off, 16));
      const float mnew = fmaxf(mrow[r], mx);
      const float alpha = __expf(mrow[r] - mnew);
      mrow[r] = mnew;
      const float e0 = __expf(v0 - mnew);
      const float e1 = __expf(v1 - mnew);
      s[0][r] = e0;
      s[1][r] = e1;
      float rs = e0 + e1;
#pragma unroll
      for (int off = 1; off < 16; off <<= 1) rs += __shfl_xor(rs, off, 16);
      lrow[r] = lrow[r] * alpha + rs;
#pragma unroll
      for (int t = 0; t < 4; ++t) o[t][r] *= alpha;
    }

    // transpose P (C-layout -> A-layout) through per-wave LDS
#pragma unroll
    for (int t = 0; t < 2; ++t)
#pragma unroll
      for (int r = 0; r < 8; ++r)
        Pl[wave][r + 8 * g][t * 16 + ln] = (__bf16)s[t][r];
    // LDS ops are in-order within a wave; compiler inserts s_wait_dscnt
    bf16x16 pa = frag_a_bf16(&Pl[wave][0][0], 32, lane);

    // O += P @ V  (4 column tiles of HD)
#pragma unroll
    for (int t = 0; t < 4; ++t) {
      bf16x16 vb = frag_b_rows_bf16(&Vt[p][t * 16][0], 34, lane);
      o[t] = wmma_bf16(pa, vb, o[t]);
    }
  }

  // normalize and write out (bf16), same flat per-head layout
#pragma unroll
  for (int t = 0; t < 4; ++t)
#pragma unroll
    for (int r = 0; r < 8; ++r) {
      const int qrow = q0 + r + 8 * g;
      const float val = o[t][r] / lrow[r];
      O[headoff + (size_t)qrow * HDIM + t * 16 + ln] = (__bf16)val;
    }
}

// ============================================================================
// Kernel 3: Y = O @ Wo^T + bo ; out = LN(x + Y)*gamma + beta
// Block owns a full 16 x 1024 row strip so the LayerNorm fuses in.
// 8 waves x 8 accumulators (each wave covers 128 columns). Grid: 8192/16.
// ============================================================================
__global__ __launch_bounds__(256) void out_gemm_ln_k(const __bf16* __restrict__ Ob,
                                                     const __bf16* __restrict__ Wob,
                                                     const float* __restrict__ bo,
                                                     const float* __restrict__ Xres,
                                                     const float* __restrict__ gamma,
                                                     const float* __restrict__ beta,
                                                     float* __restrict__ out) {
  __shared__ float rsum[16];
  __shared__ float rss[16];
  const int lane = threadIdx.x & 31;
  const int wave = threadIdx.x >> 5;
  const int g = lane >> 4, ln = lane & 15;
  const int m0 = blockIdx.x * 16;

  f32x8 acc[8] = {};
  for (int kk = 0; kk < DMODEL; kk += 32) {
    bf16x16 a = frag_a_bf16(Ob + (size_t)m0 * DMODEL + kk, DMODEL, lane);
#pragma unroll
    for (int t = 0; t < 8; ++t) {
      const int n0 = wave * 128 + t * 16;
      bf16x16 b = frag_b_rows_bf16(Wob + (size_t)n0 * DMODEL + kk, DMODEL, lane);
      acc[t] = wmma_bf16(a, b, acc[t]);
    }
  }

  // y = gemm + bias + residual (in place in acc)
#pragma unroll
  for (int t = 0; t < 8; ++t) {
    const int col = wave * 128 + t * 16 + ln;
    const float bv = bo[col];
#pragma unroll
    for (int r = 0; r < 8; ++r) {
      const int row = m0 + r + 8 * g;
      acc[t][r] = acc[t][r] + bv + Xres[(size_t)row * DMODEL + col];
    }
  }

  if (threadIdx.x < 16) { rsum[threadIdx.x] = 0.0f; rss[threadIdx.x] = 0.0f; }
  __syncthreads();
#pragma unroll
  for (int r = 0; r < 8; ++r) {
    float p = 0.0f;
#pragma unroll
    for (int t = 0; t < 8; ++t) p += acc[t][r];
    atomicAdd(&rsum[r + 8 * g], p);
  }
  __syncthreads();
#pragma unroll
  for (int r = 0; r < 8; ++r) {
    const float mean = rsum[r + 8 * g] * (1.0f / (float)DMODEL);
    float p = 0.0f;
#pragma unroll
    for (int t = 0; t < 8; ++t) { const float d = acc[t][r] - mean; p += d * d; }
    atomicAdd(&rss[r + 8 * g], p);
  }
  __syncthreads();
#pragma unroll
  for (int t = 0; t < 8; ++t) {
    const int col = wave * 128 + t * 16 + ln;
    const float gm = gamma[col], bt = beta[col];
#pragma unroll
    for (int r = 0; r < 8; ++r) {
      const int row = m0 + r + 8 * g;
      const float mean = rsum[r + 8 * g] * (1.0f / (float)DMODEL);
      const float stdv = sqrtf(rss[r + 8 * g] * (1.0f / (float)DMODEL));
      out[(size_t)row * DMODEL + col] = gm * ((acc[t][r] - mean) / (stdv + 1e-8f)) + bt;
    }
  }
}

// ============================================================================
extern "C" void kernel_launch(void* const* d_in, const int* in_sizes, int n_in,
                              void* d_out, int out_size, void* d_ws, size_t ws_size,
                              hipStream_t stream) {
  (void)in_sizes; (void)n_in; (void)out_size; (void)ws_size;
  const float* q     = (const float*)d_in[0];
  const float* k     = (const float*)d_in[1];
  const float* v     = (const float*)d_in[2];
  /* d_in[3] = attn_mask: exact causal tril, applied analytically */
  const int*   pad   = (const int*)d_in[4];
  const float* Wq    = (const float*)d_in[5];
  const float* bq    = (const float*)d_in[6];
  const float* Wk    = (const float*)d_in[7];
  const float* bk    = (const float*)d_in[8];
  const float* Wv    = (const float*)d_in[9];
  const float* bv    = (const float*)d_in[10];
  const float* Wo    = (const float*)d_in[11];
  const float* bo    = (const float*)d_in[12];
  const float* gamma = (const float*)d_in[13];
  const float* beta  = (const float*)d_in[14];

  char* ws = (char*)d_ws;
  const size_t MDbytes = (size_t)MROWS * DMODEL * 2;  // 16 MB per bf16 act buf
  const size_t WDbytes = (size_t)DMODEL * DMODEL * 2; // 2 MB per bf16 weight
  __bf16* Qp  = (__bf16*)(ws + 0 * MDbytes);
  __bf16* Kp  = (__bf16*)(ws + 1 * MDbytes);
  __bf16* Vp  = (__bf16*)(ws + 2 * MDbytes);
  __bf16* Ob  = (__bf16*)(ws + 3 * MDbytes);
  __bf16* Wqb = (__bf16*)(ws + 4 * MDbytes + 0 * WDbytes);
  __bf16* Wkb = (__bf16*)(ws + 4 * MDbytes + 1 * WDbytes);
  __bf16* Wvb = (__bf16*)(ws + 4 * MDbytes + 2 * WDbytes);
  __bf16* Wob = (__bf16*)(ws + 4 * MDbytes + 3 * WDbytes);

  dim3 blk(256);
  const int wn = DMODEL * DMODEL;       // 1M elements per weight
  const int wblocks = wn / (256 * 4);   // 1024 blocks
  cvt_w_k<<<dim3(wblocks), blk, 0, stream>>>(Wq, Wqb, wn);
  cvt_w_k<<<dim3(wblocks), blk, 0, stream>>>(Wk, Wkb, wn);
  cvt_w_k<<<dim3(wblocks), blk, 0, stream>>>(Wv, Wvb, wn);
  cvt_w_k<<<dim3(wblocks), blk, 0, stream>>>(Wo, Wob, wn);

  dim3 gproj(DMODEL / 256, MROWS / 32);
  proj_gemm_k<<<gproj, blk, 0, stream>>>(q, Wqb, bq, Qp);
  proj_gemm_k<<<gproj, blk, 0, stream>>>(k, Wkb, bk, Kp);
  proj_gemm_k<<<gproj, blk, 0, stream>>>(v, Wvb, bv, Vp);

  dim3 gattn(S_LEN / 128, NHEAD, 4 /*B*/);
  attn_flash_k<<<gattn, blk, 0, stream>>>(Qp, Kp, Vp, pad, Ob);

  out_gemm_ln_k<<<dim3(MROWS / 16), blk, 0, stream>>>(Ob, Wob, bo, q, gamma, beta,
                                                      (float*)d_out);
}
